// Seq2Seq_80994493268474
// MI455X (gfx1250) — matrix operand order repaired and checked by reading, not compiled, not run on hardware
//
#include <hip/hip_runtime.h>
#include <hip/hip_bf16.h>
#include <math.h>

// ---------------------------------------------------------------------------
// Seq2Seq (encoder LSTM + Bahdanau-attention decoder LSTM + vocab projection)
// for MI455X / gfx1250: all GEMMs via V_WMMA_F32_16X16X32_BF16 (wave32).
// ---------------------------------------------------------------------------

typedef __bf16 bf16_t;
typedef __attribute__((ext_vector_type(16))) __bf16 v16bf;
typedef __attribute__((ext_vector_type(8)))  float  v8f;

static constexpr int Bm = 16;     // batch
static constexpr int Tm = 64;     // time steps
static constexpr int Vm = 32000;  // vocab
static constexpr int Em = 256;    // embed dim
static constexpr int Hm = 1024;   // hidden dim

__device__ __forceinline__ float sigmoid_f(float x) { return 1.0f / (1.0f + expf(-x)); }

// --- WMMA fragment loaders (CDNA5 ISA 7.12.2 layouts, 16-bit A 16x32 / B 32x16)
__device__ __forceinline__ v16bf load_afrag(const bf16_t* __restrict__ A,
                                            int mrow, int K, int k0, int hiA) {
  v16bf a;
#pragma unroll
  for (int v = 0; v < 8; ++v) {
    int k = k0 + ((v < 4) ? 0 : 16) + hiA + (v & 3) * 2;
    a[2 * v]     = A[(size_t)mrow * K + k];
    a[2 * v + 1] = A[(size_t)mrow * K + k + 1];
  }
  return a;
}

__device__ __forceinline__ v16bf load_bfrag(const float* __restrict__ Bmat,
                                            int N, int k0, int kbB, int ncol) {
  v16bf b;
#pragma unroll
  for (int v = 0; v < 8; ++v) {
    int k = k0 + ((v < 4) ? 0 : 16) + (kbB >> 1) + (v & 3) * 2;
    b[2 * v]     = (bf16_t)Bmat[(size_t)k * N + ncol];
    b[2 * v + 1] = (bf16_t)Bmat[(size_t)(k + 1) * N + ncol];
  }
  return b;
}

// ---------------------------------------------------------------------------
// WMMA GEMM, 1 tile / wave (M=16 recurrent steps):
//   C[16,N] = A[16,K](bf16) * B[K,N](f32->bf16 in-register)
// 8 waves / block; K-loop in 32-chunks, last chunk peeled so the main loop
// prefetches unconditionally with an incremented pointer (no branch/mul).
// ---------------------------------------------------------------------------
__global__ __launch_bounds__(256)
void wmma_gemm16(const bf16_t* __restrict__ A,
                 const float*  __restrict__ Bmat,
                 float*        __restrict__ C,
                 int N, int K) {
  const int lane = threadIdx.x & 31;
  const int wave = threadIdx.x >> 5;
  const int n0   = (blockIdx.x * 8 + wave) * 16;
  const int mrow = lane & 15;
  const int ncol = n0 + (lane & 15);
  const int hiA  = (lane < 16) ? 0 : 8;
  const int kbB  = (lane < 16) ? 0 : 16;

  v8f acc = {0.f, 0.f, 0.f, 0.f, 0.f, 0.f, 0.f, 0.f};

  const float* pf = Bmat + (size_t)(32 + kbB) * N + ncol;
  int k0 = 0;
  for (; k0 < K - 32; k0 += 32) {
    __builtin_prefetch(pf, 0, 1);  // global_prefetch_b8, next B chunk
    pf += (size_t)32 * N;
    v16bf afrag = load_afrag(A, mrow, K, k0, hiA);
    v16bf bfrag = load_bfrag(Bmat, N, k0, kbB, ncol);
    acc = __builtin_amdgcn_wmma_f32_16x16x32_bf16(
        false, afrag, false, bfrag, (short)0, acc, false, false);
  }
  {  // peeled final chunk
    v16bf afrag = load_afrag(A, mrow, K, k0, hiA);
    v16bf bfrag = load_bfrag(Bmat, N, k0, kbB, ncol);
    acc = __builtin_amdgcn_wmma_f32_16x16x32_bf16(
        false, afrag, false, bfrag, (short)0, acc, false, false);
  }

#pragma unroll
  for (int r = 0; r < 8; ++r) {
    int mm = r + hiA;  // lanes 0-15 -> M=r, lanes 16-31 -> M=8+r
    C[(size_t)mm * N + ncol] = acc[r];
  }
}

// ---------------------------------------------------------------------------
// WMMA GEMM, 4 N-tiles / wave (big GEMMs): wave tile = 16x64, A fragment
// reused by 4 WMMAs per K-chunk. 4 waves / block cover 256 columns;
// grid = (N/256, M/16).
// ---------------------------------------------------------------------------
__global__ __launch_bounds__(128)
void wmma_gemm4(const bf16_t* __restrict__ A,
                const float*  __restrict__ Bmat,
                float*        __restrict__ C,
                int M, int N, int K) {
  const int lane = threadIdx.x & 31;
  const int wave = threadIdx.x >> 5;
  const int n0   = (blockIdx.x * 4 + wave) * 64;  // wave owns 64 columns
  const int m0   = blockIdx.y * 16;
  const int mrow = m0 + (lane & 15);
  const int nlan = lane & 15;
  const int hiA  = (lane < 16) ? 0 : 8;
  const int kbB  = (lane < 16) ? 0 : 16;

  v8f acc[4];
#pragma unroll
  for (int tIdx = 0; tIdx < 4; ++tIdx)
    acc[tIdx] = (v8f){0.f, 0.f, 0.f, 0.f, 0.f, 0.f, 0.f, 0.f};

  const float* pf0 = Bmat + (size_t)(32 + kbB) * N + n0 + nlan;
  const float* pf1 = pf0 + 32;
  int k0 = 0;
  for (; k0 < K - 32; k0 += 32) {
    __builtin_prefetch(pf0, 0, 1);
    __builtin_prefetch(pf1, 0, 1);
    pf0 += (size_t)32 * N;
    pf1 += (size_t)32 * N;
    v16bf afrag = load_afrag(A, mrow, K, k0, hiA);
#pragma unroll
    for (int tIdx = 0; tIdx < 4; ++tIdx) {
      v16bf bfrag = load_bfrag(Bmat, N, k0, kbB, n0 + tIdx * 16 + nlan);
      acc[tIdx] = __builtin_amdgcn_wmma_f32_16x16x32_bf16(
          false, afrag, false, bfrag, (short)0, acc[tIdx], false, false);
    }
  }
  {  // peeled final chunk
    v16bf afrag = load_afrag(A, mrow, K, k0, hiA);
#pragma unroll
    for (int tIdx = 0; tIdx < 4; ++tIdx) {
      v16bf bfrag = load_bfrag(Bmat, N, k0, kbB, n0 + tIdx * 16 + nlan);
      acc[tIdx] = __builtin_amdgcn_wmma_f32_16x16x32_bf16(
          false, afrag, false, bfrag, (short)0, acc[tIdx], false, false);
    }
  }

#pragma unroll
  for (int tIdx = 0; tIdx < 4; ++tIdx) {
    const int ncol = n0 + tIdx * 16 + nlan;
#pragma unroll
    for (int r = 0; r < 8; ++r) {
      int mm = m0 + r + hiA;
      C[(size_t)mm * N + ncol] = acc[tIdx][r];
    }
  }
}

// ---------------------------------------------------------------------------
// Elementwise / helper kernels
// ---------------------------------------------------------------------------
__global__ void embed_gather(const int* __restrict__ tok,
                             const float* __restrict__ emb,
                             bf16_t* __restrict__ out, int nBT) {
  int idx = blockIdx.x * blockDim.x + threadIdx.x;
  if (idx >= nBT * Em) return;
  int bt = idx / Em, e = idx % Em;
  out[idx] = (bf16_t)emb[(size_t)tok[bt] * Em + e];
}

__global__ void init_enc_state(float* __restrict__ h, float* __restrict__ c,
                               bf16_t* __restrict__ A_enc) {
  int idx = blockIdx.x * blockDim.x + threadIdx.x;
  if (idx >= Bm * Hm) return;
  int b = idx / Hm, hid = idx % Hm;
  h[idx] = 0.f;
  c[idx] = 0.f;
  A_enc[b * (Em + Hm) + Em + hid] = (bf16_t)0.f;
}

__global__ void init_dec_state(const float* __restrict__ h,
                               bf16_t* __restrict__ A_dec) {
  int idx = blockIdx.x * blockDim.x + threadIdx.x;
  if (idx >= Bm * Hm) return;
  int b = idx / Hm, hid = idx % Hm;
  const int lda = Em + 2 * Hm;
  A_dec[b * lda + Em + hid]      = (bf16_t)0.f;     // attn0 = 0
  A_dec[b * lda + Em + Hm + hid] = (bf16_t)h[idx];  // h0 = enc_h
}

__global__ void pack_x(const bf16_t* __restrict__ emb_bt,
                       bf16_t* __restrict__ Abuf, int t, int lda) {
  int idx = blockIdx.x * blockDim.x + threadIdx.x;
  if (idx >= Bm * Em) return;
  int b = idx / Em, e = idx % Em;
  Abuf[b * lda + e] = emb_bt[(b * Tm + t) * Em + e];
}

// Encoder LSTM gates + sequence-length masking + memory write
__global__ void enc_pointwise(const float* __restrict__ z,
                              const float* __restrict__ bias,
                              float* __restrict__ h, float* __restrict__ c,
                              const int* __restrict__ enc_len, int t,
                              float* __restrict__ memory,
                              bf16_t* __restrict__ memory_bf,
                              bf16_t* __restrict__ A_enc) {
  int idx = blockIdx.x * blockDim.x + threadIdx.x;
  if (idx >= Bm * Hm) return;
  int b = idx / Hm, hid = idx % Hm;
  const int zr = b * 4 * Hm;
  float gi = z[zr + hid]          + bias[hid];
  float gj = z[zr + Hm + hid]     + bias[Hm + hid];
  float gf = z[zr + 2 * Hm + hid] + bias[2 * Hm + hid];
  float go = z[zr + 3 * Hm + hid] + bias[3 * Hm + hid];
  float c_old = c[idx], h_old = h[idx];
  float c_new = c_old * sigmoid_f(gf + 1.0f) + sigmoid_f(gi) * tanhf(gj);
  float h_new = tanhf(c_new) * sigmoid_f(go);
  bool valid = t < enc_len[b];
  float hv = valid ? h_new : h_old;
  float cv = valid ? c_new : c_old;
  h[idx] = hv;
  c[idx] = cv;
  float mem = valid ? h_new : 0.f;
  memory[((size_t)b * Tm + t) * Hm + hid]    = mem;
  memory_bf[((size_t)b * Tm + t) * Hm + hid] = (bf16_t)mem;
  A_enc[b * (Em + Hm) + Em + hid] = (bf16_t)hv;
}

// Decoder LSTM gates (unconditional state update)
__global__ void dec_pointwise(const float* __restrict__ z,
                              const float* __restrict__ bias,
                              float* __restrict__ h, float* __restrict__ c,
                              bf16_t* __restrict__ A_dec,
                              bf16_t* __restrict__ h2_bf,
                              bf16_t* __restrict__ hc_cat) {
  int idx = blockIdx.x * blockDim.x + threadIdx.x;
  if (idx >= Bm * Hm) return;
  int b = idx / Hm, hid = idx % Hm;
  const int zr = b * 4 * Hm;
  float gi = z[zr + hid]          + bias[hid];
  float gj = z[zr + Hm + hid]     + bias[Hm + hid];
  float gf = z[zr + 2 * Hm + hid] + bias[2 * Hm + hid];
  float go = z[zr + 3 * Hm + hid] + bias[3 * Hm + hid];
  float c_new = c[idx] * sigmoid_f(gf + 1.0f) + sigmoid_f(gi) * tanhf(gj);
  float h_new = tanhf(c_new) * sigmoid_f(go);
  h[idx] = h_new;
  c[idx] = c_new;
  const int lda = Em + 2 * Hm;
  A_dec[b * lda + Em + Hm + hid] = (bf16_t)h_new;  // h slot for next step
  h2_bf[idx]               = (bf16_t)h_new;        // A for query GEMM
  hc_cat[b * 2 * Hm + hid] = (bf16_t)h_new;        // first half of [h2|ctx]
}

// Bahdanau scores: one block per (b,t), 256-thread reduction over H
__global__ void attn_scores(const float* __restrict__ keys,
                            const float* __restrict__ query,
                            const float* __restrict__ v_att,
                            const int* __restrict__ enc_len,
                            float* __restrict__ scores) {
  int b = blockIdx.x / Tm, t = blockIdx.x % Tm;
  int tid = threadIdx.x;
  float part = 0.f;
  for (int hh = tid; hh < Hm; hh += 256)
    part += tanhf(keys[((size_t)b * Tm + t) * Hm + hh] + query[b * Hm + hh]) * v_att[hh];
  __shared__ float red[256];
  red[tid] = part;
  __syncthreads();
  for (int s = 128; s > 0; s >>= 1) {
    if (tid < s) red[tid] += red[tid + s];
    __syncthreads();
  }
  if (tid == 0) scores[b * Tm + t] = (t < enc_len[b]) ? red[0] : -1e9f;
}

// softmax over T + context = align @ memory ; writes ctx half of hc_cat
__global__ void softmax_context(const float* __restrict__ scores,
                                const float* __restrict__ memory,
                                bf16_t* __restrict__ hc_cat) {
  int b = blockIdx.x, tid = threadIdx.x;
  __shared__ float al[Tm];
  if (tid == 0) {
    float mx = -1e30f;
    for (int t = 0; t < Tm; ++t) mx = fmaxf(mx, scores[b * Tm + t]);
    float s = 0.f;
    for (int t = 0; t < Tm; ++t) { float e = expf(scores[b * Tm + t] - mx); al[t] = e; s += e; }
    float inv = 1.f / s;
    for (int t = 0; t < Tm; ++t) al[t] *= inv;
  }
  __syncthreads();
  for (int hh = tid; hh < Hm; hh += 256) {
    float ctx = 0.f;
    for (int t = 0; t < Tm; ++t) ctx += al[t] * memory[((size_t)b * Tm + t) * Hm + hh];
    hc_cat[b * 2 * Hm + Hm + hh] = (bf16_t)ctx;
  }
}

// attn2 -> next-step A slot + masked decoder output row (bf16 for final GEMM)
__global__ void attn_post(const float* __restrict__ attn2,
                          const int* __restrict__ dec_len, int t,
                          bf16_t* __restrict__ A_dec,
                          bf16_t* __restrict__ dec_out_bf) {
  int idx = blockIdx.x * blockDim.x + threadIdx.x;
  if (idx >= Bm * Hm) return;
  int b = idx / Hm, hid = idx % Hm;
  float a2 = attn2[idx];
  const int lda = Em + 2 * Hm;
  A_dec[b * lda + Em + hid] = (bf16_t)a2;
  bool fin = t >= dec_len[b];
  dec_out_bf[((size_t)b * Tm + t) * Hm + hid] = fin ? (bf16_t)0.f : (bf16_t)a2;
}

// ---------------------------------------------------------------------------
// Host-side orchestration
// ---------------------------------------------------------------------------
extern "C" void kernel_launch(void* const* d_in, const int* in_sizes, int n_in,
                              void* d_out, int out_size, void* d_ws, size_t ws_size,
                              hipStream_t stream) {
  (void)in_sizes; (void)n_in; (void)out_size; (void)ws_size;

  const int*   enc_in    = (const int*)d_in[0];
  const int*   dec_in    = (const int*)d_in[1];
  const int*   enc_len   = (const int*)d_in[2];
  const int*   dec_len   = (const int*)d_in[3];
  const float* embedding = (const float*)d_in[4];
  const float* enc_W     = (const float*)d_in[5];
  const float* enc_b     = (const float*)d_in[6];
  const float* dec_W     = (const float*)d_in[7];
  const float* dec_b     = (const float*)d_in[8];
  const float* Wm        = (const float*)d_in[9];
  const float* Wq        = (const float*)d_in[10];
  const float* v_att     = (const float*)d_in[11];
  const float* attn_W    = (const float*)d_in[12];
  const float* out_W     = (const float*)d_in[13];
  float* logits = (float*)d_out;

  // workspace carve-up (256B aligned)
  char* p = (char*)d_ws;
  auto alloc = [&](size_t bytes) { char* r = p; p += (bytes + 255) & ~(size_t)255; return r; };
  bf16_t* q_emb      = (bf16_t*)alloc((size_t)Bm * Tm * Em * sizeof(bf16_t));
  bf16_t* r_emb      = (bf16_t*)alloc((size_t)Bm * Tm * Em * sizeof(bf16_t));
  bf16_t* A_enc      = (bf16_t*)alloc((size_t)Bm * (Em + Hm) * sizeof(bf16_t));
  bf16_t* A_dec      = (bf16_t*)alloc((size_t)Bm * (Em + 2 * Hm) * sizeof(bf16_t));
  float*  h_st       = (float*) alloc((size_t)Bm * Hm * sizeof(float));
  float*  c_st       = (float*) alloc((size_t)Bm * Hm * sizeof(float));
  float*  z_buf      = (float*) alloc((size_t)Bm * 4 * Hm * sizeof(float));
  float*  memory     = (float*) alloc((size_t)Bm * Tm * Hm * sizeof(float));
  bf16_t* memory_bf  = (bf16_t*)alloc((size_t)Bm * Tm * Hm * sizeof(bf16_t));
  float*  keys       = (float*) alloc((size_t)Bm * Tm * Hm * sizeof(float));
  bf16_t* h2_bf      = (bf16_t*)alloc((size_t)Bm * Hm * sizeof(bf16_t));
  float*  query      = (float*) alloc((size_t)Bm * Hm * sizeof(float));
  float*  scores     = (float*) alloc((size_t)Bm * Tm * sizeof(float));
  bf16_t* hc_cat     = (bf16_t*)alloc((size_t)Bm * 2 * Hm * sizeof(bf16_t));
  float*  attn2      = (float*) alloc((size_t)Bm * Hm * sizeof(float));
  bf16_t* dec_out_bf = (bf16_t*)alloc((size_t)Bm * Tm * Hm * sizeof(bf16_t));

  const dim3 blk256(256), blk128(128);
  const int nBH_blocks  = (Bm * Hm + 255) / 256;       // 64
  const int nBE_blocks  = (Bm * Em + 255) / 256;       // 16
  const int nEmb_blocks = (Bm * Tm * Em + 255) / 256;  // 1024

  // 1) embedding gathers (f32 -> bf16 in-register)
  embed_gather<<<nEmb_blocks, blk256, 0, stream>>>(enc_in, embedding, q_emb, Bm * Tm);
  embed_gather<<<nEmb_blocks, blk256, 0, stream>>>(dec_in, embedding, r_emb, Bm * Tm);

  // 2) encoder: 64 sequential steps of [16 x 1280] @ [1280 x 4096]
  init_enc_state<<<nBH_blocks, blk256, 0, stream>>>(h_st, c_st, A_enc);
  for (int t = 0; t < Tm; ++t) {
    pack_x<<<nBE_blocks, blk256, 0, stream>>>(q_emb, A_enc, t, Em + Hm);
    wmma_gemm16<<<dim3(4 * Hm / 128, 1), blk256, 0, stream>>>(
        A_enc, enc_W, z_buf, 4 * Hm, Em + Hm);
    enc_pointwise<<<nBH_blocks, blk256, 0, stream>>>(
        z_buf, enc_b, h_st, c_st, enc_len, t, memory, memory_bf, A_enc);
  }

  // 3) keys = memory @ Wm   (1024 x 1024 x 1024), 16x64 wave tiles
  wmma_gemm4<<<dim3(Hm / 256, (Bm * Tm) / 16), blk128, 0, stream>>>(
      memory_bf, Wm, keys, Bm * Tm, Hm, Hm);

  // 4) decoder with attention (state h_st/c_st carried from encoder)
  init_dec_state<<<nBH_blocks, blk256, 0, stream>>>(h_st, A_dec);
  for (int t = 0; t < Tm; ++t) {
    pack_x<<<nBE_blocks, blk256, 0, stream>>>(r_emb, A_dec, t, Em + 2 * Hm);
    wmma_gemm16<<<dim3(4 * Hm / 128, 1), blk256, 0, stream>>>(
        A_dec, dec_W, z_buf, 4 * Hm, Em + 2 * Hm);
    dec_pointwise<<<nBH_blocks, blk256, 0, stream>>>(
        z_buf, dec_b, h_st, c_st, A_dec, h2_bf, hc_cat);
    wmma_gemm16<<<dim3(Hm / 128, 1), blk256, 0, stream>>>(
        h2_bf, Wq, query, Hm, Hm);
    attn_scores<<<Bm * Tm, blk256, 0, stream>>>(keys, query, v_att, enc_len, scores);
    softmax_context<<<Bm, blk256, 0, stream>>>(scores, memory, hc_cat);
    wmma_gemm16<<<dim3(Hm / 128, 1), blk256, 0, stream>>>(
        hc_cat, attn_W, attn2, Hm, 2 * Hm);
    attn_post<<<nBH_blocks, blk256, 0, stream>>>(attn2, dec_len, t, A_dec, dec_out_bf);
  }

  // 5) logits = dec_out @ out_kernel   (1024 x 32000 x 1024) — dominant GEMM
  wmma_gemm4<<<dim3(Vm / 256, (Bm * Tm) / 16), blk128, 0, stream>>>(
      dec_out_bf, out_W, logits, Bm * Tm, Vm, Hm);
}
